// MultiHeadAttention_84112639525199
// MI455X (gfx1250) — matrix operand order repaired and checked by reading, not compiled, not run on hardware
//
#include <hip/hip_runtime.h>

// ---------------------------------------------------------------------------
// Shaw relative-position MHA fused for MI455X (gfx1250, wave32, bf16 WMMA).
//   B=2, S=1500, HID=1024, NH=16, HD=64, KCLIP=64 (129 distinct distances)
// Pipeline:
//   memset QKV pad region
//   prep kernels    : W/Wfc/pe -> bf16, transposed so every WMMA B-fragment
//                     is two 16B vector loads
//   proj_kernel x3  : X @ W + b -> Q,K bf16 [B,NH,S',HD]; V bf16 [B,NH,HD,S']
//                     (32x64 tile per wave: 2 A-frags share each B-frag)
//   attn_kernel     : per (b, head, 16-row q tile), one wave32:
//                       Qp = Qtile @ pe_k^T (16x129 -> LDS)
//                       pass A: row max of (QK^T + Shaw)/8
//                       pass B: p=exp(s-m); l+=p; wsum[dist]+=p (LDS atomics);
//                               O += P @ V (P staged via LDS)
//                       out = O/l + (wsum/l) @ pe_v -> hidden[B,S,HID]
//   fc_kernel       : hidden @ W_fc + b -> out f32 (32x64 tile per wave)
// ---------------------------------------------------------------------------

#define BATCH 2
#define S_LEN 1500
#define SP    1504     // padded sequence (multiple of 32)
#define HIDN  1024
#define NH    16
#define HD    64
#define KCLIP 64
#define NREL  129
#define PK_R  144      // pe_k rows padded
#define QP_LD 144      // Qp LDS row stride
#define WS_LD 160      // wsum LDS row stride / pe_v K padding (5 x 32)
#define NROW  (BATCH * S_LEN)   // 3000

typedef __attribute__((ext_vector_type(16))) __bf16 v16bf;
typedef __attribute__((ext_vector_type(8)))  __bf16 v8bf;
typedef __attribute__((ext_vector_type(8)))  float  v8f;

static __device__ __forceinline__ v8f wmma_bf16(v16bf a, v16bf b, v8f c) {
  return __builtin_amdgcn_wmma_f32_16x16x32_bf16(false, a, false, b, (short)0, c,
                                                 false, false);
}
static __device__ __forceinline__ v8f vzero8() {
  v8f z = {0.f, 0.f, 0.f, 0.f, 0.f, 0.f, 0.f, 0.f};
  return z;
}

// A fragment (16x32, 16-bit): lane half h, element e -> K = (e<8?e:e+8)+8h.
static __device__ __forceinline__ v16bf a_frag_f32(const float* row, int h, float msk) {
  const float* r1 = row + 8 * h;
  float4 a0 = *(const float4*)(r1);
  float4 a1 = *(const float4*)(r1 + 4);
  float4 a2 = *(const float4*)(r1 + 16);
  float4 a3 = *(const float4*)(r1 + 20);
  float xf[16] = {a0.x, a0.y, a0.z, a0.w, a1.x, a1.y, a1.z, a1.w,
                  a2.x, a2.y, a2.z, a2.w, a3.x, a3.y, a3.z, a3.w};
  v16bf a;
#pragma unroll
  for (int e = 0; e < 16; ++e) a[e] = (__bf16)(xf[e] * msk);
  return a;
}
static __device__ __forceinline__ v16bf a_frag_bf16(const __bf16* row, int h) {
  v8bf lo = *(const v8bf*)(row + 8 * h);
  v8bf hi = *(const v8bf*)(row + 16 + 8 * h);
  v16bf a;
#pragma unroll
  for (int e = 0; e < 8; ++e) { a[e] = lo[e]; a[e + 8] = hi[e]; }
  return a;
}
// B fragment (32x16, 16-bit): lane half h, element e -> K = (e<8?e:e+8)+8h.
static __device__ __forceinline__ v16bf b_frag_bf16(const __bf16* colbase, int h) {
  v8bf lo = *(const v8bf*)(colbase + 8 * h);
  v8bf hi = *(const v8bf*)(colbase + 16 + 8 * h);
  v16bf b;
#pragma unroll
  for (int e = 0; e < 8; ++e) { b[e] = lo[e]; b[e + 8] = hi[e]; }
  return b;
}

// ---------------------------------------------------------------------------
// Weight / PE preparation (bf16, transposed for contiguous B-fragments)
// ---------------------------------------------------------------------------
__global__ __launch_bounds__(256)
void prep_w_perhead(const float* __restrict__ W,   // [NH, HIDN, HD]
                    __bf16* __restrict__ Wt)       // [1024(c), HIDN(k)]
{
  int idx = blockIdx.x * 256 + threadIdx.x;        // 1024*1024
  int c = idx >> 10, k = idx & 1023;
  Wt[idx] = (__bf16)W[((size_t)(c >> 6) * HIDN + k) * HD + (c & 63)];
}
__global__ __launch_bounds__(256)
void prep_w_fc(const float* __restrict__ W,        // [HIDN, HIDN]
               __bf16* __restrict__ Wt)            // [1024(c), HIDN(k)]
{
  int idx = blockIdx.x * 256 + threadIdx.x;
  int c = idx >> 10, k = idx & 1023;
  Wt[idx] = (__bf16)W[(size_t)k * HIDN + c];
}
__global__ __launch_bounds__(256)
void prep_pe(const float* __restrict__ pk,         // [NREL, HD]
             const float* __restrict__ pv,         // [NREL, HD]
             __bf16* __restrict__ pekb,            // [PK_R, HD] rows>=129 zero
             __bf16* __restrict__ pevT)            // [HD, WS_LD] cols>=129 zero
{
  int idx = blockIdx.x * 256 + threadIdx.x;
  if (idx < PK_R * HD) {
    int r = idx / HD;
    pekb[idx] = (__bf16)((r < NREL) ? pk[idx] : 0.f);
  }
  if (idx < HD * WS_LD) {
    int d = idx / WS_LD, r = idx % WS_LD;
    pevT[idx] = (__bf16)((r < NREL) ? pv[r * HD + d] : 0.f);
  }
}

// ---------------------------------------------------------------------------
// QKV projection: one wave -> 32x64 output tile (2 A-frags share each B-frag).
// vT=0: out[b,n,s,d] (Q,K).  vT=1: out[b,n,d,s] (V transposed).
// ---------------------------------------------------------------------------
__global__ __launch_bounds__(32)
void proj_kernel(const float* __restrict__ x,      // [NROW, HIDN]
                 const __bf16* __restrict__ Wt,    // [1024(c), HIDN(k)]
                 const float* __restrict__ bias,   // [1024]
                 __bf16* __restrict__ out, int vT)
{
  const int lane = threadIdx.x, half = lane >> 4, l16 = lane & 15;
  const int m0 = blockIdx.x * 32;
  const int c0 = blockIdx.y * 64;

  const int mA0 = m0 + l16, mA1 = m0 + 16 + l16;
  const bool ok0 = (mA0 < NROW), ok1 = (mA1 < NROW);
  const float msk0 = ok0 ? 1.f : 0.f, msk1 = ok1 ? 1.f : 0.f;
  const float* xr0 = x + (size_t)(ok0 ? mA0 : NROW - 1) * HIDN;
  const float* xr1 = x + (size_t)(ok1 ? mA1 : NROW - 1) * HIDN;

  v8f acc[8];
#pragma unroll
  for (int i = 0; i < 8; ++i) acc[i] = vzero8();

  for (int kb = 0; kb < HIDN; kb += 32) {
    if (kb + 32 < HIDN) {                      // warm L2->WGP path one step ahead
      __builtin_prefetch(xr0 + kb + 32, 0, 3);
      __builtin_prefetch(xr1 + kb + 32, 0, 3);
    }
    v16bf a0 = a_frag_f32(xr0 + kb, half, msk0);
    v16bf a1 = a_frag_f32(xr1 + kb, half, msk1);
#pragma unroll
    for (int nt = 0; nt < 4; ++nt) {
      v16bf b = b_frag_bf16(Wt + (size_t)(c0 + nt * 16 + l16) * HIDN + kb, half);
      acc[nt]     = wmma_bf16(a0, b, acc[nt]);
      acc[4 + nt] = wmma_bf16(a1, b, acc[4 + nt]);
    }
  }
#pragma unroll
  for (int nt = 0; nt < 4; ++nt) {
    const int colc = c0 + nt * 16 + l16;
    const int head = colc >> 6, dcol = colc & 63;
    const float bv = bias[colc];
#pragma unroll
    for (int mt = 0; mt < 2; ++mt) {
#pragma unroll
      for (int r = 0; r < 8; ++r) {
        int m = m0 + mt * 16 + r + 8 * half;
        if (m < NROW) {
          int bi = (m >= S_LEN) ? 1 : 0;       // BATCH == 2
          int s = m - bi * S_LEN;
          size_t o = vT ? (((size_t)bi * NH + head) * HD + dcol) * SP + s
                        : (((size_t)bi * NH + head) * SP + s) * HD + dcol;
          out[o] = (__bf16)(acc[mt * 4 + nt][r] + bv);
        }
      }
    }
  }
}

// ---------------------------------------------------------------------------
// Fused attention: one wave per (b, head, 16-row q tile).
// ---------------------------------------------------------------------------
__global__ __launch_bounds__(32)
void attn_kernel(const __bf16* __restrict__ Qb,    // [B,NH,SP,HD] (pad rows 0)
                 const __bf16* __restrict__ Kb,    // [B,NH,SP,HD] (pad rows 0)
                 const __bf16* __restrict__ Vt,    // [B,NH,HD,SP] (pad cols 0)
                 const __bf16* __restrict__ pekb,  // [PK_R, HD]
                 const __bf16* __restrict__ pevT,  // [HD, WS_LD]
                 float* __restrict__ hidden)       // [B, S, HIDN]
{
  __shared__ float Qp[16 * QP_LD];
  __shared__ float Wsum[16 * WS_LD];
  __shared__ float Pt[16 * 32];
  __shared__ float lrow[16];

  const int lane = threadIdx.x, half = lane >> 4, l16 = lane & 15;
  const int q0 = blockIdx.x * 16;
  const int headn = blockIdx.y;
  const int bidx = blockIdx.z;

  const size_t bn = (size_t)bidx * NH + headn;
  const __bf16* Qh = Qb + bn * SP * HD;
  const __bf16* Kh = Kb + bn * SP * HD;
  const __bf16* Vh = Vt + bn * HD * SP;

  for (int i = lane; i < 16 * WS_LD; i += 32) Wsum[i] = 0.f;

  // Q A-fragments (K = 0..31, 32..63 of HD); padded rows are zero.
  const __bf16* qrow = Qh + (size_t)(q0 + l16) * HD;
  v16bf qa[2];
#pragma unroll
  for (int kb2 = 0; kb2 < 2; ++kb2) qa[kb2] = a_frag_bf16(qrow + kb2 * 32, half);

  // Qp = Qtile(16x64) @ pe_k^T(64x129) -> LDS
  for (int nt = 0; nt < 9; ++nt) {
    const int rcol = nt * 16 + l16;          // relative-distance column (<144)
    v8f acc = vzero8();
#pragma unroll
    for (int kb2 = 0; kb2 < 2; ++kb2) {
      v16bf b = b_frag_bf16(pekb + (size_t)rcol * HD + kb2 * 32, half);
      acc = wmma_bf16(qa[kb2], b, acc);
    }
#pragma unroll
    for (int r = 0; r < 8; ++r) Qp[(r + 8 * half) * QP_LD + rcol] = acc[r];
  }
  __syncthreads();

  const float scale = 0.125f;                // 1/sqrt(HD)
  const int NKT = SP / 16;                   // 94 k tiles

  float rm[8], ls[8];
#pragma unroll
  for (int r = 0; r < 8; ++r) { rm[r] = -1e30f; ls[r] = 0.f; }

  // ---- pass A: row max ----
  for (int j = 0; j < NKT; ++j) {
    const int kg = j * 16 + l16;             // B-layout: N = lane&15
    const __bf16* krow = Kh + (size_t)kg * HD;
    if (j + 1 < NKT) __builtin_prefetch(krow + 16 * HD, 0, 3);
    v8f sc = vzero8();
#pragma unroll
    for (int kb2 = 0; kb2 < 2; ++kb2)
      sc = wmma_bf16(qa[kb2], b_frag_bf16(krow + kb2 * 32, half), sc);
#pragma unroll
    for (int r = 0; r < 8; ++r) {
      int M = r + 8 * half;
      int dist = kg - (q0 + M);
      dist = (dist < -KCLIP) ? -KCLIP : (dist > KCLIP ? KCLIP : dist);
      float s = (sc[r] + Qp[M * QP_LD + dist + KCLIP]) * scale;
      s = (kg < S_LEN) ? s : -1e30f;
      rm[r] = fmaxf(rm[r], s);
    }
  }
#pragma unroll
  for (int r = 0; r < 8; ++r)
    for (int o = 8; o > 0; o >>= 1)
      rm[r] = fmaxf(rm[r], __shfl_xor(rm[r], o, 32));

  // ---- pass B: exp/accumulate + O += P @ V ----
  v8f O[4];
#pragma unroll
  for (int nt = 0; nt < 4; ++nt) O[nt] = vzero8();

  for (int j32 = 0; j32 < NKT / 2; ++j32) {
#pragma unroll
    for (int sub = 0; sub < 2; ++sub) {
      const int j = j32 * 2 + sub;
      const int kg = j * 16 + l16;
      const __bf16* krow = Kh + (size_t)kg * HD;
      if (j + 1 < NKT) __builtin_prefetch(krow + 16 * HD, 0, 3);
      v8f sc = vzero8();
#pragma unroll
      for (int kb2 = 0; kb2 < 2; ++kb2)
        sc = wmma_bf16(qa[kb2], b_frag_bf16(krow + kb2 * 32, half), sc);
#pragma unroll
      for (int r = 0; r < 8; ++r) {
        int M = r + 8 * half;
        int dist = kg - (q0 + M);
        dist = (dist < -KCLIP) ? -KCLIP : (dist > KCLIP ? KCLIP : dist);
        float s = (sc[r] + Qp[M * QP_LD + dist + KCLIP]) * scale;
        s = (kg < S_LEN) ? s : -1e30f;       // masked -> p == 0
        float p = __expf(s - rm[r]);
        ls[r] += p;
        atomicAdd(&Wsum[M * WS_LD + dist + KCLIP], p);
        Pt[M * 32 + sub * 16 + l16] = p;
      }
    }
    // P(16x32) @ V(32x64)
    v16bf pa = a_frag_f32(Pt + l16 * 32, half, 1.f);
#pragma unroll
    for (int nt = 0; nt < 4; ++nt) {
      v16bf b = b_frag_bf16(Vh + (size_t)(nt * 16 + l16) * SP + j32 * 32, half);
      O[nt] = wmma_bf16(pa, b, O[nt]);
    }
  }

  // ---- softmax denominators ----
#pragma unroll
  for (int r = 0; r < 8; ++r)
    for (int o = 8; o > 0; o >>= 1)
      ls[r] += __shfl_xor(ls[r], o, 32);
  if (l16 == 0) {
#pragma unroll
    for (int r = 0; r < 8; ++r) lrow[r + 8 * half] = ls[r];
  }
  __syncthreads();
  for (int i = lane; i < 16 * WS_LD; i += 32)
    Wsum[i] = Wsum[i] / lrow[i / WS_LD];
  __syncthreads();

  // ---- out = O/l + wsum_norm(16x160) @ pe_v(160x64) ----
#pragma unroll
  for (int nt = 0; nt < 4; ++nt) {
    v8f wb = vzero8();
    for (int kb5 = 0; kb5 < 5; ++kb5) {
      v16bf a = a_frag_f32(Wsum + l16 * WS_LD + kb5 * 32, half, 1.f);
      v16bf b = b_frag_bf16(pevT + (size_t)(nt * 16 + l16) * WS_LD + kb5 * 32, half);
      wb = wmma_bf16(a, b, wb);
    }
#pragma unroll
    for (int r = 0; r < 8; ++r) {
      int M = r + 8 * half;
      int q = q0 + M;
      if (q < S_LEN) {
        float val = O[nt][r] / lrow[M] + wb[r];
        hidden[((size_t)bidx * S_LEN + q) * HIDN + headn * HD + nt * 16 + l16] = val;
      }
    }
  }
}

// ---------------------------------------------------------------------------
// Final projection: hidden @ W_fc + b_fc -> out f32 (32x64 tile per wave)
// ---------------------------------------------------------------------------
__global__ __launch_bounds__(32)
void fc_kernel(const float* __restrict__ hidden,   // [NROW, HIDN]
               const __bf16* __restrict__ Wt,      // [1024(c), HIDN(k)]
               const float* __restrict__ bfc,      // [1024]
               float* __restrict__ out)            // [NROW, HIDN]
{
  const int lane = threadIdx.x, half = lane >> 4, l16 = lane & 15;
  const int m0 = blockIdx.x * 32;
  const int c0 = blockIdx.y * 64;

  const int mA0 = m0 + l16, mA1 = m0 + 16 + l16;
  const bool ok0 = (mA0 < NROW), ok1 = (mA1 < NROW);
  const float msk0 = ok0 ? 1.f : 0.f, msk1 = ok1 ? 1.f : 0.f;
  const float* xr0 = hidden + (size_t)(ok0 ? mA0 : NROW - 1) * HIDN;
  const float* xr1 = hidden + (size_t)(ok1 ? mA1 : NROW - 1) * HIDN;

  v8f acc[8];
#pragma unroll
  for (int i = 0; i < 8; ++i) acc[i] = vzero8();

  for (int kb = 0; kb < HIDN; kb += 32) {
    if (kb + 32 < HIDN) {
      __builtin_prefetch(xr0 + kb + 32, 0, 3);
      __builtin_prefetch(xr1 + kb + 32, 0, 3);
    }
    v16bf a0 = a_frag_f32(xr0 + kb, half, msk0);
    v16bf a1 = a_frag_f32(xr1 + kb, half, msk1);
#pragma unroll
    for (int nt = 0; nt < 4; ++nt) {
      v16bf b = b_frag_bf16(Wt + (size_t)(c0 + nt * 16 + l16) * HIDN + kb, half);
      acc[nt]     = wmma_bf16(a0, b, acc[nt]);
      acc[4 + nt] = wmma_bf16(a1, b, acc[4 + nt]);
    }
  }
#pragma unroll
  for (int nt = 0; nt < 4; ++nt) {
    const int colc = c0 + nt * 16 + l16;
    const float bv = bfc[colc];
#pragma unroll
    for (int mt = 0; mt < 2; ++mt) {
#pragma unroll
      for (int r = 0; r < 8; ++r) {
        int m = m0 + mt * 16 + r + 8 * half;
        if (m < NROW)
          out[(size_t)m * HIDN + colc] = acc[mt * 4 + nt][r] + bv;
      }
    }
  }
}

// ---------------------------------------------------------------------------
extern "C" void kernel_launch(void* const* d_in, const int* in_sizes, int n_in,
                              void* d_out, int out_size, void* d_ws, size_t ws_size,
                              hipStream_t stream) {
  (void)in_sizes; (void)n_in; (void)out_size; (void)ws_size;
  const float* query = (const float*)d_in[0];
  const float* key   = (const float*)d_in[1];
  const float* value = (const float*)d_in[2];
  const float* Wq    = (const float*)d_in[3];
  const float* bq    = (const float*)d_in[4];
  const float* Wk    = (const float*)d_in[5];
  const float* bk    = (const float*)d_in[6];
  const float* Wv    = (const float*)d_in[7];
  const float* bv    = (const float*)d_in[8];
  const float* pe_k  = (const float*)d_in[9];
  const float* pe_v  = (const float*)d_in[10];
  const float* W_fc  = (const float*)d_in[11];
  const float* b_fc  = (const float*)d_in[12];
  float* out = (float*)d_out;

  // ---- workspace layout (~39.2 MB) ----
  const size_t qkvBytes = (size_t)BATCH * NH * SP * HD * sizeof(__bf16); // 6,160,384
  const size_t hidBytes = (size_t)BATCH * S_LEN * HIDN * sizeof(float);  // 12,288,000
  const size_t wtBytes  = (size_t)HIDN * HIDN * sizeof(__bf16);          // 2,097,152
  char* ws = (char*)d_ws;
  __bf16* Qb   = (__bf16*)(ws);
  __bf16* Kb   = (__bf16*)(ws + qkvBytes);
  __bf16* Vt   = (__bf16*)(ws + 2 * qkvBytes);
  float* hidden = (float*)(ws + 3 * qkvBytes);
  __bf16* Wqt  = (__bf16*)(ws + 3 * qkvBytes + hidBytes);
  __bf16* Wkt  = (__bf16*)(ws + 3 * qkvBytes + hidBytes + wtBytes);
  __bf16* Wvt  = (__bf16*)(ws + 3 * qkvBytes + hidBytes + 2 * wtBytes);
  __bf16* Wfct = (__bf16*)(ws + 3 * qkvBytes + hidBytes + 3 * wtBytes);
  __bf16* pekb = (__bf16*)(ws + 3 * qkvBytes + hidBytes + 4 * wtBytes);
  __bf16* pevT = (__bf16*)(ws + 3 * qkvBytes + hidBytes + 4 * wtBytes
                              + (size_t)PK_R * HD * sizeof(__bf16));

  // zero QKV buffers (covers the 4 padded rows/cols read by attention)
  hipMemsetAsync(ws, 0, 3 * qkvBytes, stream);

  dim3 b256(256);
  prep_w_perhead<<<dim3((HIDN * HIDN) / 256), b256, 0, stream>>>(Wq, Wqt);
  prep_w_perhead<<<dim3((HIDN * HIDN) / 256), b256, 0, stream>>>(Wk, Wkt);
  prep_w_perhead<<<dim3((HIDN * HIDN) / 256), b256, 0, stream>>>(Wv, Wvt);
  prep_w_fc<<<dim3((HIDN * HIDN) / 256), b256, 0, stream>>>(W_fc, Wfct);
  prep_pe<<<dim3((HD * WS_LD + 255) / 256), b256, 0, stream>>>(pe_k, pe_v, pekb, pevT);

  dim3 blk(32);
  dim3 gproj((NROW + 31) / 32, (NH * HD) / 64);
  proj_kernel<<<gproj, blk, 0, stream>>>(query, Wqt, bq, Qb, 0);
  proj_kernel<<<gproj, blk, 0, stream>>>(key,   Wkt, bk, Kb, 0);
  proj_kernel<<<gproj, blk, 0, stream>>>(value, Wvt, bv, Vt, 1);

  dim3 gattn(SP / 16, NH, BATCH);
  attn_kernel<<<gattn, blk, 0, stream>>>(Qb, Kb, Vt, pekb, pevT, hidden);

  dim3 gfc((NROW + 31) / 32, HIDN / 64);
  fc_kernel<<<gfc, blk, 0, stream>>>(hidden, Wfct, b_fc, out);
}
